// MambaTrendBackbone_11948599018275
// MI455X (gfx1250) — matrix operand order repaired and checked by reading, not compiled, not run on hardware
//
#include <hip/hip_runtime.h>
#include <hip/hip_bf16.h>

typedef __attribute__((ext_vector_type(16))) __bf16 v16bf;
typedef __attribute__((ext_vector_type(8)))  float  v8f;

#define D_MODEL 256
#define SEQ     2048
#define BATCH   64
#define HORIZON 90
#define N_LAYERS 4
#define ROWS (BATCH*SEQ)   // 131072 (b,s) rows

// dynamic LDS for scan kernel: 2 weight operand sets (A,C) + ping-pong state
#define SCAN_LDS_BYTES (2*65536*2 /*wts bf16*/ + 2*8*32*16*2 /*state bf16*/)  // 278528

// ---------- swizzle helpers ----------
// A-operand (16x32 bf16, ISA 7.12.2): element e(0..15) per lane, lane = m + 16*hh
//   e in 0..7 : k = e + 8*hh ; e in 8..15: k = 16 + (e-8) + 8*hh
__device__ __forceinline__ void a_swz(int kk, int& hh, int& e) {
    if (kk < 16) { hh = kk >> 3; e = kk & 7; }
    else         { int t = kk - 16; hh = t >> 3; e = 8 + (t & 7); }
}

// flat index into the A-operand-swizzled bf16 activation buffer for (row, d)
__device__ __forceinline__ size_t hbf_idx(int row, int d) {
    int mtile = row >> 4, mr = row & 15;
    int c = d >> 5, kk = d & 31;
    int hh, e; a_swz(kk, hh, e);
    return ((size_t)(mtile*8 + c)*32 + (mr + 16*hh))*16 + e;
}

// ---------- weight pre-swizzle: A/B/C (4 layers, 256x256 f32) -> bf16 B-operand tiles ----------
// B-operand 32x16 bf16: lane = nn + 16*hh, element e with (hh,e) = a_swz(kk) (same k order as A). B_op[k][n] = W[n][k].
__global__ void prep_weights(const float* A, const float* B, const float* C, __bf16* w) {
    int id = blockIdx.x*blockDim.x + threadIdx.x;
    if (id >= N_LAYERS*3*D_MODEL*D_MODEL) return;
    int k   = id & 255;
    int n   = (id >> 8) & 255;
    int lm  = id >> 16;          // l*3 + mat
    int mat = lm % 3;
    int l   = lm / 3;
    const float* src = (mat == 0) ? A : (mat == 1) ? B : C;
    float v = src[(size_t)l*65536 + n*256 + k];
    int ntile = n >> 4, nn = n & 15;
    int c = k >> 5,  kk = k & 31;
    int hh, e; a_swz(kk, hh, e);
    size_t idx = ((((size_t)lm*16 + ntile)*8 + c)*32 + (nn + 16*hh))*16 + e;
    w[idx] = (__bf16)v;
}

// ---------- input projection: h = x @ Wp^T + bp  (K=5, memory-bound) ----------
__global__ void proj_kernel(const float* __restrict__ x, const float* __restrict__ Wp,
                            const float* __restrict__ bp, float* h, __bf16* hbf) {
    int id = blockIdx.x*blockDim.x + threadIdx.x;   // ROWS*256 threads
    int d = id & 255, row = id >> 8;
    const float* xr = x + (size_t)row*5;
    const float* wr = Wp + d*5;
    float v = bp[d];
    #pragma unroll
    for (int i = 0; i < 5; ++i) v += xr[i]*wr[i];
    h[id] = v;
    hbf[hbf_idx(row, d)] = (__bf16)v;
}

// ---------- xB = h @ B_l^T : bf16 WMMA GEMM, M=131072 N=256 K=256 ----------
// One mtile per wave: activation A-operands (8 x v16bf) stay in registers and are
// read from HBM exactly once; the 128 KB weight set streams from L2 (resident).
__global__ void gemm_xb(const __bf16* __restrict__ hbf, const __bf16* __restrict__ wB,
                        float* __restrict__ xb) {
    int lane = threadIdx.x & 31;
    int wid  = blockIdx.x*(blockDim.x >> 5) + (threadIdx.x >> 5);
    int nwav = gridDim.x*(blockDim.x >> 5);
    int hl = lane >> 4, nl = lane & 15;
    const int MTILES = ROWS/16;                    // 8192
    for (int mtile = wid; mtile < MTILES; mtile += nwav) {
        // load the full 16x256 activation strip (A-operands) once
        v16bf a[8];
        #pragma unroll
        for (int c = 0; c < 8; ++c)
            a[c] = *(const v16bf*)(hbf + ((size_t)(mtile*8 + c)*32 + lane)*16);
        // sweep all 16 output ntiles, reusing the register-resident activations
        #pragma unroll 2
        for (int ntile = 0; ntile < 16; ++ntile) {
            v8f acc = {};
            #pragma unroll
            for (int c = 0; c < 8; ++c) {
                v16bf b = *(const v16bf*)(wB + ((size_t)(ntile*8 + c)*32 + lane)*16);
                acc = __builtin_amdgcn_wmma_f32_16x16x32_bf16(false, a[c], false, b,
                                                              (short)0, acc, false, false);
            }
            #pragma unroll
            for (int r = 0; r < 8; ++r) {
                int row = mtile*16 + r + 8*hl;
                xb[(size_t)row*256 + ntile*16 + nl] = acc[r];
            }
        }
    }
}

// ---------- sequential scan: h_t = tanh(h_{t-1}@A^T + xB_t); y_t = h_t@C^T ----------
// One workgroup per 16-batch tile (WMMA M). A/C operand tiles preloaded to LDS
// (256 KB of the WGP's 320 KB); state in LDS bf16 ping-pong; xB for step t+1
// prefetched into registers behind step t's WMMA chain. One barrier per step.
__global__ void __launch_bounds__(256, 1)
scan_kernel(const __bf16* __restrict__ wA, const __bf16* __restrict__ wC,
            float* __restrict__ xby) {
    extern __shared__ char smem[];
    __bf16* wlds = (__bf16*)smem;                        // [2][16][8][32][16] = 256 KB
    __bf16* hbuf = (__bf16*)(smem + 2*65536*2);          // [2][8*32*16]       =  16 KB

    int lane = threadIdx.x & 31;
    int w    = threadIdx.x >> 5;                   // 0..7
    int btile = blockIdx.x;                        // 0..3
    int hl = lane >> 4, nl = lane & 15;

    // preload weight operand tiles (global -> LDS, 16B chunks)
    for (int i = threadIdx.x; i < 65536/8; i += 256) {
        ((uint4*)wlds)[i]         = ((const uint4*)wA)[i];
        ((uint4*)(wlds+65536))[i] = ((const uint4*)wC)[i];
    }
    // zero initial state (both buffers)
    for (int i = threadIdx.x; i < 2*8*32*16; i += 256) hbuf[i] = (__bf16)0.0f;
    __syncthreads();

    int j0 = 2*w, j1 = 2*w + 1;
    int hh0, e0, hh1, e1;
    { int kk0 = (j0 & 1)*16 + nl; a_swz(kk0, hh0, e0);
      int kk1 = (j1 & 1)*16 + nl; a_swz(kk1, hh1, e1); }
    int c0t = j0 >> 1, c1t = j1 >> 1;

    const __bf16* wAl = wlds;
    const __bf16* wCl = wlds + 65536;

    // prefetch xB accumulators for t = 0
    v8f nxt0, nxt1;
    #pragma unroll
    for (int r = 0; r < 8; ++r) {
        int m = r + 8*hl;
        size_t base = ((size_t)(btile*16 + m)*SEQ + 0)*256;
        nxt0[r] = xby[base + j0*16 + nl];
        nxt1[r] = xby[base + j1*16 + nl];
    }

    int p = 0;
    for (int t = 0; t < SEQ; ++t) {
        v8f acc0 = nxt0, acc1 = nxt1;
        // prefetch next step's xB tiles (overlaps with this step's WMMA chain)
        int tn = (t + 1 < SEQ) ? (t + 1) : (SEQ - 1);
        #pragma unroll
        for (int r = 0; r < 8; ++r) {
            int m = r + 8*hl;
            size_t base = ((size_t)(btile*16 + m)*SEQ + tn)*256;
            nxt0[r] = xby[base + j0*16 + nl];
            nxt1[r] = xby[base + j1*16 + nl];
        }
        // h_prev @ A^T (operands all from LDS)
        #pragma unroll
        for (int c = 0; c < 8; ++c) {
            v16bf a  = *(const v16bf*)&hbuf[(p*4096) + (c*32 + lane)*16];
            v16bf b0 = *(const v16bf*)&wAl[((j0*8 + c)*32 + lane)*16];
            acc0 = __builtin_amdgcn_wmma_f32_16x16x32_bf16(false, a, false, b0,
                                                           (short)0, acc0, false, false);
            v16bf b1 = *(const v16bf*)&wAl[((j1*8 + c)*32 + lane)*16];
            acc1 = __builtin_amdgcn_wmma_f32_16x16x32_bf16(false, a, false, b1,
                                                           (short)0, acc1, false, false);
        }
        // tanh, scatter new state to LDS (A-operand layout)
        int q = p ^ 1;
        #pragma unroll
        for (int r = 0; r < 8; ++r) {
            float v0 = tanhf(acc0[r]);
            float v1 = tanhf(acc1[r]);
            int m = r + 8*hl;
            hbuf[(q*4096) + (c0t*32 + m + 16*hh0)*16 + e0] = (__bf16)v0;
            hbuf[(q*4096) + (c1t*32 + m + 16*hh1)*16 + e1] = (__bf16)v1;
        }
        __syncthreads();
        // y_t = h_t @ C^T ; overwrite xB slot in place (read in a prior step)
        v8f y0 = {}, y1 = {};
        #pragma unroll
        for (int c = 0; c < 8; ++c) {
            v16bf a  = *(const v16bf*)&hbuf[(q*4096) + (c*32 + lane)*16];
            v16bf b0 = *(const v16bf*)&wCl[((j0*8 + c)*32 + lane)*16];
            y0 = __builtin_amdgcn_wmma_f32_16x16x32_bf16(false, a, false, b0,
                                                         (short)0, y0, false, false);
            v16bf b1 = *(const v16bf*)&wCl[((j1*8 + c)*32 + lane)*16];
            y1 = __builtin_amdgcn_wmma_f32_16x16x32_bf16(false, a, false, b1,
                                                         (short)0, y1, false, false);
        }
        #pragma unroll
        for (int r = 0; r < 8; ++r) {
            int m = r + 8*hl;
            size_t base = ((size_t)(btile*16 + m)*SEQ + t)*256;
            xby[base + j0*16 + nl] = y0[r];
            xby[base + j1*16 + nl] = y1[r];
        }
        p = q;
    }
}

// ---------- residual + LayerNorm; also emit next-layer bf16 swizzled activations ----------
__global__ void ln_kernel(const float* __restrict__ y, float* __restrict__ h,
                          __bf16* __restrict__ hbf,
                          const float* __restrict__ g, const float* __restrict__ bta) {
    int lane = threadIdx.x & 31;
    int w    = threadIdx.x >> 5;
    int row  = blockIdx.x*8 + w;                   // one wave per row
    const float* yr = y + (size_t)row*256;
    float* hr = h + (size_t)row*256;
    float zv[8];
    float s = 0.f, s2 = 0.f;
    #pragma unroll
    for (int i = 0; i < 8; ++i) {
        int d = i*32 + lane;
        float v = yr[d] + hr[d];
        zv[i] = v; s += v; s2 += v*v;
    }
    #pragma unroll
    for (int m = 16; m >= 1; m >>= 1) {
        s  += __shfl_xor(s,  m, 32);
        s2 += __shfl_xor(s2, m, 32);
    }
    float mean = s * (1.0f/256.0f);
    float var  = s2 * (1.0f/256.0f) - mean*mean;
    float inv  = rsqrtf(var + 1e-5f);
    #pragma unroll
    for (int i = 0; i < 8; ++i) {
        int d = i*32 + lane;
        float o = (zv[i] - mean)*inv*g[d] + bta[d];
        hr[d] = o;
        hbf[hbf_idx(row, d)] = (__bf16)o;
    }
}

// ---------- heads on last timestep: 3 GEMVs + gelu-MLP softmax mixture ----------
__global__ void head_kernel(const float* __restrict__ h,
                            const float* Ws, const float* bs,
                            const float* Wm, const float* bm,
                            const float* Wl, const float* bl,
                            const float* W1, const float* b1,
                            const float* W2, const float* b2,
                            float* __restrict__ out) {
    __shared__ float hf[256];
    __shared__ float u[64];
    __shared__ float z[3];
    int b = blockIdx.x, tid = threadIdx.x;
    hf[tid] = h[((size_t)b*SEQ + (SEQ-1))*256 + tid];
    __syncthreads();
    if (tid < 64) {
        float a = b1[tid];
        for (int d = 0; d < 256; ++d) a += hf[d]*W1[tid*256 + d];
        u[tid] = 0.5f*a*(1.0f + erff(a*0.70710678118654752f));   // exact gelu
    }
    __syncthreads();
    if (tid < 3) {
        float a = b2[tid];
        for (int j = 0; j < 64; ++j) a += u[j]*W2[tid*64 + j];
        z[tid] = a;
    }
    __syncthreads();
    float zmax = fmaxf(z[0], fmaxf(z[1], z[2]));
    float ea = expf(z[0]-zmax), eb = expf(z[1]-zmax), ec = expf(z[2]-zmax);
    float inv = 1.0f/(ea+eb+ec);
    float w0 = ea*inv, w1 = eb*inv, w2 = ec*inv;
    if (tid < HORIZON) {
        float ps = bs[tid], pm = bm[tid], pl = bl[tid];
        for (int d = 0; d < 256; ++d) {
            float hv = hf[d];
            ps += hv*Ws[tid*256 + d];
            pm += hv*Wm[tid*256 + d];
            pl += hv*Wl[tid*256 + d];
        }
        out[b*HORIZON + tid] = w0*ps + w1*pm + w2*pl;
    }
}

extern "C" void kernel_launch(void* const* d_in, const int* in_sizes, int n_in,
                              void* d_out, int out_size, void* d_ws, size_t ws_size,
                              hipStream_t stream) {
    (void)in_sizes; (void)n_in; (void)out_size; (void)ws_size;
    const float* x   = (const float*)d_in[0];
    const float* Wp  = (const float*)d_in[1];
    const float* bp  = (const float*)d_in[2];
    const float* A   = (const float*)d_in[3];
    const float* B   = (const float*)d_in[4];
    const float* C   = (const float*)d_in[5];
    const float* lng = (const float*)d_in[6];
    const float* lnb = (const float*)d_in[7];
    const float* Ws  = (const float*)d_in[8];
    const float* bs  = (const float*)d_in[9];
    const float* Wm  = (const float*)d_in[10];
    const float* bm  = (const float*)d_in[11];
    const float* Wl  = (const float*)d_in[12];
    const float* bl  = (const float*)d_in[13];
    const float* W1  = (const float*)d_in[14];
    const float* b1  = (const float*)d_in[15];
    const float* W2  = (const float*)d_in[16];
    const float* b2  = (const float*)d_in[17];

    // workspace layout (~337 MB)
    char*   ws    = (char*)d_ws;
    float*  h_f32 = (float*)ws;                        // 64*2048*256*4   = 134217728 B
    __bf16* hbf   = (__bf16*)(ws + 134217728ull);      // bf16 swizzled   =  67108864 B
    float*  xby   = (float*)(ws + 201326592ull);       // xB / y shared   = 134217728 B
    __bf16* wswz  = (__bf16*)(ws + 335544320ull);      // swizzled wts    =   1572864 B

    // allow 272 KB dynamic LDS for the scan kernel (WGP has 320 KB)
    (void)hipFuncSetAttribute((const void*)scan_kernel,
                              hipFuncAttributeMaxDynamicSharedMemorySize,
                              SCAN_LDS_BYTES);

    prep_weights<<<(N_LAYERS*3*D_MODEL*D_MODEL + 255)/256, 256, 0, stream>>>(A, B, C, wswz);
    proj_kernel<<<ROWS, 256, 0, stream>>>(x, Wp, bp, h_f32, hbf);

    for (int l = 0; l < N_LAYERS; ++l) {
        const __bf16* wAl = wswz + (size_t)(l*3 + 0)*65536;
        const __bf16* wBl = wswz + (size_t)(l*3 + 1)*65536;
        const __bf16* wCl = wswz + (size_t)(l*3 + 2)*65536;
        gemm_xb<<<1024, 256, 0, stream>>>(hbf, wBl, xby);
        scan_kernel<<<4, 256, SCAN_LDS_BYTES, stream>>>(wAl, wCl, xby);
        ln_kernel<<<ROWS/8, 256, 0, stream>>>(xby, h_f32, hbf, lng + l*256, lnb + l*256);
    }

    head_kernel<<<BATCH, 256, 0, stream>>>(h_f32, Ws, bs, Wm, bm, Wl, bl,
                                           W1, b1, W2, b2, (float*)d_out);
}
